// OmniDynamicConv1d_12378095747481
// MI455X (gfx1250) — matrix operand order repaired and checked by reading, not compiled, not run on hardware
//
#include <hip/hip_runtime.h>
#include <math.h>

typedef __attribute__((ext_vector_type(16))) __bf16 v16bf;
typedef __attribute__((ext_vector_type(8)))  float  v8f;
typedef __attribute__((ext_vector_type(4)))  unsigned int uint4v;
typedef __attribute__((ext_vector_type(4)))  unsigned int u32x4;
typedef __attribute__((ext_vector_type(8)))  int i32x8;
typedef __attribute__((ext_vector_type(4)))  int i32x4;

#define B_SZ  32
#define C_IN  256
#define C_OUT 256
#define KW    7
#define N_WT  8
#define SQ    8
#define L_LEN 4096

// workspace layout (bytes)
#define WS_POOLED 0         // f32 [32][256]       = 32768
#define WS_KR     32768     // f32 [32][7]  (pad)  -> 1024
#define WS_INC    33792     // f32 [32][256]       = 32768
#define WS_OUTC   66560     // f32 [32][256]       = 32768
#define WS_NUMS   99328     // f32 [32][8]         -> 1024
#define WS_WMIX   100352    // bf16 [32][7][256][256] = 29360128 bytes

#if __has_builtin(__builtin_amdgcn_tensor_load_to_lds)
#define HAVE_TDM 1
#else
#define HAVE_TDM 0
#endif

static __device__ __forceinline__ unsigned short f2bf(float f) {
  union { float f; unsigned int u; } v; v.f = f;
  unsigned int r = v.u + 0x7FFFu + ((v.u >> 16) & 1u);   // round-to-nearest-even
  return (unsigned short)(r >> 16);
}

struct F2 { uint4v lo, hi; };
static __device__ __forceinline__ v16bf make_frag(uint4v lo, uint4v hi) {
  F2 t; t.lo = lo; t.hi = hi;
  return __builtin_bit_cast(v16bf, t);
}

// ---------------- Stage 1: mean pool over L ----------------
__global__ __launch_bounds__(256) void pool_kernel(const float* __restrict__ x,
                                                   float* __restrict__ pooled) {
  const int bc = blockIdx.x;                 // b*256 + c
  const float* p = x + (size_t)bc * L_LEN;
  float s = 0.0f;
  for (int l = threadIdx.x; l < L_LEN; l += 256) s += p[l];
  __shared__ float red[256];
  red[threadIdx.x] = s;
  __syncthreads();
  for (int st = 128; st > 0; st >>= 1) {
    if (threadIdx.x < st) red[threadIdx.x] += red[threadIdx.x + st];
    __syncthreads();
  }
  if (threadIdx.x == 0) pooled[bc] = red[0] * (1.0f / (float)L_LEN);
}

// ---------------- Stage 2: squeeze MLP + gating heads ----------------
__global__ __launch_bounds__(256) void gates_kernel(
    const float* __restrict__ pooled,
    const float* __restrict__ w1, const float* __restrict__ b1,
    const float* __restrict__ wk, const float* __restrict__ wi,
    const float* __restrict__ wo, const float* __restrict__ wn,
    float* __restrict__ g_kr, float* __restrict__ g_inc,
    float* __restrict__ g_outc, float* __restrict__ g_nums) {
  const int b = blockIdx.x;
  const int t = threadIdx.x;
  __shared__ float sp[C_IN];
  __shared__ float sh[SQ];
  __shared__ float se[N_WT];
  sp[t] = pooled[b * C_IN + t];
  __syncthreads();
  if (t < SQ) {
    float a = b1[t];
    for (int c = 0; c < C_IN; ++c) a += sp[c] * w1[c * SQ + t];
    sh[t] = 0.5f * a * (1.0f + erff(a * 0.70710678118654752f));   // exact GELU
  }
  __syncthreads();
  float h[SQ];
  #pragma unroll
  for (int j = 0; j < SQ; ++j) h[j] = sh[j];
  {
    float s = 0.0f;
    #pragma unroll
    for (int j = 0; j < SQ; ++j) s += h[j] * wi[j * C_IN + t];
    g_inc[b * C_IN + t] = 1.0f / (1.0f + expf(-s));
  }
  {
    float s = 0.0f;
    #pragma unroll
    for (int j = 0; j < SQ; ++j) s += h[j] * wo[j * C_OUT + t];
    g_outc[b * C_OUT + t] = 1.0f / (1.0f + expf(-s));
  }
  if (t < KW) {
    float s = 0.0f;
    #pragma unroll
    for (int j = 0; j < SQ; ++j) s += h[j] * wk[j * KW + t];
    g_kr[b * KW + t] = 1.0f / (1.0f + expf(-s));
  }
  if (t < N_WT) {
    float s = 0.0f;
    #pragma unroll
    for (int j = 0; j < SQ; ++j) s += h[j] * wn[j * N_WT + t];
    se[t] = s;
  }
  __syncthreads();
  if (t < N_WT) {
    float mx = se[0];
    for (int n = 1; n < N_WT; ++n) mx = fmaxf(mx, se[n]);
    float tot = 0.0f;
    for (int n = 0; n < N_WT; ++n) tot += expf(se[n] - mx);
    g_nums[b * N_WT + t] = expf(se[t] - mx) / tot;
  }
}

// ---------------- Stage 3: per-sample mixed weights (bf16, k-major) ----------------
// wmix[b][k][o][i] = outc[b,o]*inc[b,i]*kr[b,k]*sum_n nums[b,n]*cw[o,i,k,n]
__global__ __launch_bounds__(256) void mix_kernel(
    const float* __restrict__ cw,
    const float* __restrict__ g_kr, const float* __restrict__ g_inc,
    const float* __restrict__ g_outc, const float* __restrict__ g_nums,
    unsigned short* __restrict__ wmix) {
  const int blk = blockIdx.x;               // b*56 + k*8 + oc
  const int b   = blk / 56;
  const int rr  = blk - b * 56;
  const int k   = rr >> 3;
  const int oc  = rr & 7;
  const int i   = threadIdx.x;

  float nm[N_WT];
  #pragma unroll
  for (int n = 0; n < N_WT; ++n) nm[n] = g_nums[b * N_WT + n];
  const float kin = g_kr[b * KW + k] * g_inc[b * C_IN + i];

  for (int oo = 0; oo < 32; ++oo) {
    const int o = oc * 32 + oo;
    const float* p = cw + (((size_t)o * C_IN + i) * KW + k) * N_WT;
    float s = 0.0f;
    #pragma unroll
    for (int n = 0; n < N_WT; ++n) s += nm[n] * p[n];
    const float w = s * kin * g_outc[b * C_OUT + o];
    wmix[(((size_t)b * KW + k) * C_OUT + o) * C_IN + i] = f2bf(w);
  }
}

// ---------------- Stage 4: per-sample conv as WMMA GEMM ----------------
// Out[b,o,l] = sum_{i,k} wmix[b,k,o,i] * x[b,i,l+k-3]
// Block: 256 thr = 8 waves; tile 64(o) x 256(l); each wave owns 32 l-columns.
// A tiles staged by the Tensor Data Mover (one 3D TENSOR_LOAD_TO_LDS per i-step),
// overlapped with manual f32->bf16 transposed staging of the x window.
__global__ __launch_bounds__(256) void conv_wmma(
    const float* __restrict__ x, const unsigned short* __restrict__ wmix,
    float* __restrict__ out) {
  __shared__ alignas(16) unsigned short Xt[264 * 32];       // [lcol][ii], 64B rows
  __shared__ alignas(16) unsigned short At[KW * 64 * 32];   // [k][o][ii],  64B rows

  const int tid  = threadIdx.x;
  const int wv   = tid >> 5;
  const int lane = tid & 31;
  const int g    = (lane >> 4) & 1;
  const int lr   = lane & 15;
  const int l0   = blockIdx.x * 256;
  const int o0   = blockIdx.y * 64;
  const int b    = blockIdx.z;

  const float* xb = x + (size_t)b * C_IN * L_LEN;
  const unsigned short* wb = wmix + (size_t)b * KW * C_OUT * C_IN;

  v8f acc[4][2];
  #pragma unroll
  for (int ms = 0; ms < 4; ++ms)
    #pragma unroll
    for (int ns = 0; ns < 2; ++ns)
      #pragma unroll
      for (int r = 0; r < 8; ++r) acc[ms][ns][r] = 0.0f;

#if HAVE_TDM
  // Tensor DMA descriptor (uniform): tile X=32 i (64B), Y=64 o (stride 256),
  // Z=7 taps (stride 65536); contiguous into At. workgroup_mask=0 (no cluster).
  const unsigned lds_at = (unsigned)(size_t)(&At[0]);
  i32x8 g1;
  g1[0] = 1 << 16;                 // data_size = 2 bytes
  g1[1] = (int)(256u << 16);       // tensor_dim0 = 256
  g1[2] = (int)(256u << 16);       // tensor_dim1 = 256
  g1[3] = (int)(32u << 16);        // tile_dim0 = 32
  g1[4] = (int)(64u | (7u << 16)); // tile_dim1 = 64, tile_dim2 = 7
  g1[5] = 256;                     // tensor_dim0_stride = 256
  g1[6] = 0;                       // dim0_stride hi | dim1_stride lo16 (65536 -> 0)
  g1[7] = 1;                       // tensor_dim1_stride = 65536 (hi bits)
  i32x4 g2; g2[0] = 7; g2[1] = 0; g2[2] = 0; g2[3] = 0;   // tensor_dim2 = 7
  i32x4 g3; g3[0] = 0; g3[1] = 0; g3[2] = 0; g3[3] = 0;
  i32x8 g4;                         // extra descriptor group (6-arg form): unused -> 0
  #pragma unroll
  for (int q = 0; q < 8; ++q) g4[q] = 0;
#endif

  for (int ib = 0; ib < C_IN; ib += 32) {
    __syncthreads();   // previous iteration's reads of Xt/At are done

#if HAVE_TDM
    if (wv == 0) {
      unsigned long long gaddr =
          (unsigned long long)(size_t)(wb + (size_t)o0 * C_IN + ib);
      u32x4 g0;
      g0[0] = 1u;                                    // count = 1 valid descriptor
      g0[1] = lds_at;                                // LDS dest
      g0[2] = (unsigned)gaddr;                       // global addr lo
      g0[3] = (unsigned)(gaddr >> 32) | (2u << 30);  // addr hi | type=2 (image)
      __builtin_amdgcn_tensor_load_to_lds(g0, g1, g2, g3, g4, 0);
    }
#else
    for (int idx = tid; idx < KW * 64 * 4; idx += 256) {
      const int k   = idx >> 8;
      const int r   = idx & 255;
      const int row = r >> 2;
      const int c   = r & 3;
      uint4v v = *(const uint4v*)(wb + ((size_t)(k * C_OUT + o0 + row)) * C_IN + ib + c * 8);
      *(uint4v*)(&At[(k * 64 + row) * 32 + c * 8]) = v;
    }
#endif

    // Stage x window [l0-3, l0+259) for 32 channels, transposed, f32->bf16
    // (overlaps with the TDM transfer above)
    for (int idx = tid; idx < 32 * 262; idx += 256) {
      const int ii = idx / 262;
      const int lc = idx - ii * 262;
      const int gl = l0 - 3 + lc;
      float v = 0.0f;
      if (gl >= 0 && gl < L_LEN) v = xb[(size_t)(ib + ii) * L_LEN + gl];
      Xt[lc * 32 + ii] = f2bf(v);
    }

#if HAVE_TDM
    if (wv == 0) __builtin_amdgcn_s_wait_tensorcnt(0);
#endif
    __syncthreads();   // Xt staged by all waves; At landed via TDM

    const int n0 = wv * 32 + lr;          // first of this lane's two column slots
    #pragma unroll
    for (int k = 0; k < KW; ++k) {
      // B fragments: 32(K=ii) x 16(N=l) bf16 — contiguous k = 16*g + s (same as A), 2x b128 each
      const unsigned short* brow0 = &Xt[(n0 + k) * 32];
      const unsigned short* brow1 = &Xt[(n0 + 16 + k) * 32];
      v16bf bf0 = make_frag(*(const uint4v*)(brow0 + g * 16),
                            *(const uint4v*)(brow0 + g * 16 + 8));
      v16bf bf1 = make_frag(*(const uint4v*)(brow1 + g * 16),
                            *(const uint4v*)(brow1 + g * 16 + 8));
      #pragma unroll
      for (int ms = 0; ms < 4; ++ms) {
        // A fragment: 16(M=o) x 32(K=ii) bf16 — reused by two WMMAs
        const unsigned short* arow = &At[(k * 64 + ms * 16 + lr) * 32];
        v16bf af = make_frag(*(const uint4v*)(arow + g * 16),
                             *(const uint4v*)(arow + g * 16 + 8));
        acc[ms][0] = __builtin_amdgcn_wmma_f32_16x16x32_bf16(
            false, af, false, bf0, (short)0, acc[ms][0], false, false);
        acc[ms][1] = __builtin_amdgcn_wmma_f32_16x16x32_bf16(
            false, af, false, bf1, (short)0, acc[ms][1], false, false);
      }
    }
  }

  // C/D layout: VGPR r -> M = r + 8*g, N = lr
  float* ob = out + ((size_t)b * C_OUT + o0) * L_LEN + l0;
  #pragma unroll
  for (int ms = 0; ms < 4; ++ms)
    #pragma unroll
    for (int ns = 0; ns < 2; ++ns)
      #pragma unroll
      for (int r = 0; r < 8; ++r)
        ob[(size_t)(ms * 16 + r + 8 * g) * L_LEN + wv * 32 + ns * 16 + lr] =
            acc[ms][ns][r];
}

extern "C" void kernel_launch(void* const* d_in, const int* in_sizes, int n_in,
                              void* d_out, int out_size, void* d_ws, size_t ws_size,
                              hipStream_t stream) {
  const float* x  = (const float*)d_in[0];
  const float* cw = (const float*)d_in[1];
  const float* w1 = (const float*)d_in[2];
  const float* b1 = (const float*)d_in[3];
  const float* wk = (const float*)d_in[4];
  const float* wi = (const float*)d_in[5];
  const float* wo = (const float*)d_in[6];
  const float* wn = (const float*)d_in[7];
  float* out = (float*)d_out;
  char* ws = (char*)d_ws;

  float* pooled = (float*)(ws + WS_POOLED);
  float* g_kr   = (float*)(ws + WS_KR);
  float* g_inc  = (float*)(ws + WS_INC);
  float* g_outc = (float*)(ws + WS_OUTC);
  float* g_nums = (float*)(ws + WS_NUMS);
  unsigned short* wmix = (unsigned short*)(ws + WS_WMIX);

  pool_kernel<<<dim3(B_SZ * C_IN), dim3(256), 0, stream>>>(x, pooled);
  gates_kernel<<<dim3(B_SZ), dim3(256), 0, stream>>>(pooled, w1, b1, wk, wi, wo, wn,
                                                     g_kr, g_inc, g_outc, g_nums);
  mix_kernel<<<dim3(B_SZ * KW * 8), dim3(256), 0, stream>>>(cw, g_kr, g_inc, g_outc,
                                                            g_nums, wmix);
  conv_wmma<<<dim3(L_LEN / 256, C_OUT / 64, B_SZ), dim3(256), 0, stream>>>(x, wmix, out);
}
